// AutoEncoderV4_65927747994196
// MI455X (gfx1250) — matrix operand order repaired and checked by reading, not compiled, not run on hardware
//
#include <hip/hip_runtime.h>
#include <hip/hip_bf16.h>
#include <stdint.h>

// ---------------------------------------------------------------------------
// CDNA5 (gfx1250, wave32) implementation of AutoEncoderV4.
// Heavy compute (per-point MLP 3->64->128->256->512 over 81920 points,
// ~28 GFLOP) runs through v_wmma_f32_16x16x32_f16 in a fused kernel that
// keeps activations in LDS (f16) and GEMM outputs in LDS (f32) for exact
// LayerNorm statistics. Weights are converted once per call to f16, N-major
// (K-contiguous) so each lane's B-fragment is one contiguous 32B load.
// Layer-0 inputs are staged into LDS with GLOBAL_LOAD_ASYNC_TO_LDS_B128
// (ASYNCcnt / s_wait_asynccnt) when the toolchain exposes the builtins.
// ---------------------------------------------------------------------------

typedef __attribute__((ext_vector_type(16))) _Float16 v16h;
typedef __attribute__((ext_vector_type(8)))  _Float16 v8h;
typedef __attribute__((ext_vector_type(8)))  float    v8f;

#define LN_EPS 1e-3f

constexpr int MT   = 64;              // rows (points) per block tile
constexpr int STRA = 520;             // f16 activation LDS stride (pad vs 512)
constexpr int STRY = 516;             // f32 gemm-out LDS stride
constexpr int LDS_A_BYTES = MT * STRA * 2;            // 66560
constexpr int LDS_Y_BYTES = MT * STRY * 4;            // 132096
constexpr int LDS_BYTES   = LDS_A_BYTES + LDS_Y_BYTES; // ~194 KB < 320 KB/WGP

// ---------------------------------------------------------------------------
// CDNA5 async Memory->LDS path (guarded: falls back to plain staging)
// ---------------------------------------------------------------------------
#if defined(__gfx1250__) && __has_builtin(__builtin_amdgcn_global_load_async_to_lds_b128)
#define HAVE_ASYNC_LDS 1
#endif

#ifdef HAVE_ASYNC_LDS
typedef int i32x4 __attribute__((vector_size(16)));   // b128 payload type
typedef __attribute__((address_space(1))) i32x4* g_b128_p;   // global (AS1)
typedef __attribute__((address_space(3))) i32x4* l_b128_p;   // LDS (AS3)
__device__ __forceinline__ void async_b128_to_lds(const void* gsrc, void* lds)
{
  // generic->as1: numeric identity; generic->as3: low 32 bits are the LDS
  // byte offset (ISA 10.2 aperture mapping).
  __builtin_amdgcn_global_load_async_to_lds_b128(
      (g_b128_p)(uintptr_t)gsrc,
      (l_b128_p)(uint32_t)(uintptr_t)lds,
      0, 0);
}
__device__ __forceinline__ void wait_asynccnt0()
{
#if __has_builtin(__builtin_amdgcn_s_wait_asynccnt)
  __builtin_amdgcn_s_wait_asynccnt(0);
#else
  asm volatile("s_wait_asynccnt 0x0" ::: "memory");
#endif
}
#endif

// ---------------------------------------------------------------------------
// small helpers
// ---------------------------------------------------------------------------
__device__ __forceinline__ float block_sum(float v, float* red) {
  const int t = threadIdx.x;
  red[t] = v;
  __syncthreads();
  #pragma unroll
  for (int s = 128; s > 0; s >>= 1) {
    if (t < s) red[t] += red[t + s];
    __syncthreads();
  }
  float r = red[0];
  __syncthreads();
  return r;
}

// ---------------------------------------------------------------------------
// Fused backbone building blocks
// ---------------------------------------------------------------------------
// GEMM: bufA (MT x K, f16, LDS)  @  Wt (N x K, f16, global, K-contiguous)
//   -> bufY (MT x N, f32, LDS), + bias.
// 8 waves split the 4 x (N/16) grid of 16x16 WMMA tiles.
__device__ __forceinline__ void gemm_layer(const _Float16* bufA, float* bufY,
    const _Float16* __restrict__ wt, const float* __restrict__ bias,
    int K, int N, int lane, int wave)
{
  const int ntiles = N >> 4;
  const int tiles  = 4 * ntiles;          // MT/16 == 4
  const int lr = lane & 15;
  const int lh = lane >> 4;
  for (int tix = wave; tix < tiles; tix += 8) {
    const int mi = tix & 3;
    const int ni = tix >> 2;
    v8f acc = {};
    // A fragment (16-bit A 16x32): lane lr = row, lh selects
    // K-range {lh*16 .. lh*16+15} (contiguous, matching B).
    const _Float16* aRow = bufA + (mi * 16 + lr) * STRA + lh * 16;
    // B fragment: Wt is N-major; lane lr = output column, lh*16 K-offset,
    // 16 contiguous halves per lane per k-step.
    const _Float16* bRow = wt + (size_t)(ni * 16 + lr) * K + lh * 16;
    for (int kk = 0; kk < K; kk += 32) {
      v8h alo = *(const v8h*)(aRow + kk);
      v8h ahi = *(const v8h*)(aRow + kk + 8);
      v16h a = __builtin_shufflevector(alo, ahi,
               0,1,2,3,4,5,6,7,8,9,10,11,12,13,14,15);
      v16h b = *(const v16h*)(bRow + kk);
      if (kk + 32 < K) __builtin_prefetch(bRow + kk + 32, 0, 1);
      acc = __builtin_amdgcn_wmma_f32_16x16x32_f16(
              false, a, false, b, (short)0, acc, false, false);
    }
    // C layout: lane lr = column, VGPR v = row (v + 8*lh) within tile.
    const int n  = ni * 16 + lr;
    const float bs = bias[n];
    const int m0 = mi * 16 + lh * 8;
    #pragma unroll
    for (int v = 0; v < 8; ++v)
      bufY[(m0 + v) * STRY + n] = acc[v] + bs;
  }
  __syncthreads();
}

// LayerNorm(+gamma,beta)+ReLU over feature dim, per row; one wave per row
// (8 rows per wave), 32-lane shuffle reduction; writes f16 activations.
__device__ __forceinline__ void ln_relu(const float* bufY, _Float16* bufA,
    const float* __restrict__ gam, const float* __restrict__ bet,
    int N, int lane, int wave)
{
  const float invN = 1.0f / (float)N;
  for (int r = wave; r < MT; r += 8) {
    const float* y = bufY + r * STRY;
    float s = 0.f, ss = 0.f;
    for (int n = lane; n < N; n += 32) { float v = y[n]; s += v; ss += v * v; }
    #pragma unroll
    for (int off = 16; off > 0; off >>= 1) {
      s  += __shfl_xor(s,  off);
      ss += __shfl_xor(ss, off);
    }
    const float mean = s * invN;
    const float var  = ss * invN - mean * mean;
    const float rs   = rsqrtf(var + LN_EPS);
    _Float16* hx = bufA + r * STRA;
    for (int n = lane; n < N; n += 32) {
      float h = (y[n] - mean) * rs * gam[n] + bet[n];
      hx[n] = (_Float16)fmaxf(h, 0.f);
    }
  }
  __syncthreads();
}

// ---------------------------------------------------------------------------
// Fused per-point MLP: layer0 (3->64 scalar) then 3 WMMA layers, LN+ReLU
// between, final block max-pool into gmax[sample][512] via int atomicMax
// (values are >= 0 after ReLU, so float order == int order on bit patterns).
// ---------------------------------------------------------------------------
__global__ __launch_bounds__(256)
void k_backbone(const float* __restrict__ src,          // [Mtotal][3]
  const float* __restrict__ w0,  const float* __restrict__ b0,
  const float* __restrict__ g0,  const float* __restrict__ be0,
  const _Float16* __restrict__ wt1, const float* __restrict__ b1,
  const float* __restrict__ g1,  const float* __restrict__ be1,
  const _Float16* __restrict__ wt2, const float* __restrict__ b2,
  const float* __restrict__ g2,  const float* __restrict__ be2,
  const _Float16* __restrict__ wt3, const float* __restrict__ b3,
  const float* __restrict__ g3,  const float* __restrict__ be3,
  float* __restrict__ gmax)
{
  extern __shared__ char smem[];
  _Float16* bufA = (_Float16*)smem;
  float*    bufY = (float*)(smem + LDS_A_BYTES);
  const int lane = threadIdx.x & 31;
  const int wave = threadIdx.x >> 5;
  const int rowBase = blockIdx.x * MT;

  // Stage this block's 64x3 f32 points (768 B, contiguous) into LDS.
  // bufA is free until ln_relu() writes it, so reuse its head as the stage.
  float* pts = (float*)bufA;
  const float* srcBlk = src + (size_t)rowBase * 3;
#ifdef HAVE_ASYNC_LDS
  if (threadIdx.x < 48)        // 48 x 16B = 768B, issued by waves 0..1
    async_b128_to_lds((const char*)srcBlk + threadIdx.x * 16,
                      (char*)pts + threadIdx.x * 16);
  wait_asynccnt0();
  __syncthreads();
#else
  for (int i = threadIdx.x; i < MT * 3; i += 256) pts[i] = srcBlk[i];
  __syncthreads();
#endif

  // layer0: 3 -> 64 (scalar; 3 MACs per output)
  for (int e = threadIdx.x; e < MT * 64; e += 256) {
    const int r = e >> 6, c = e & 63;
    const float x = pts[r * 3], y = pts[r * 3 + 1], z = pts[r * 3 + 2];
    bufY[r * STRY + c] = x * w0[c] + y * w0[64 + c] + z * w0[128 + c] + b0[c];
  }
  __syncthreads();
  ln_relu(bufY, bufA, g0, be0, 64, lane, wave);

  gemm_layer(bufA, bufY, wt1, b1, 64, 128, lane, wave);
  ln_relu(bufY, bufA, g1, be1, 128, lane, wave);
  gemm_layer(bufA, bufY, wt2, b2, 128, 256, lane, wave);
  ln_relu(bufY, bufA, g2, be2, 256, lane, wave);
  gemm_layer(bufA, bufY, wt3, b3, 256, 512, lane, wave);
  ln_relu(bufY, bufA, g3, be3, 512, lane, wave);

  // max over the 64 rows of this block (all belong to one sample: 512 rows
  // per sample, MT=64 divides 512), then one atomic per column.
  const int sample = rowBase >> 9;
  for (int n = threadIdx.x; n < 512; n += 256) {
    float m = 0.f;
    #pragma unroll 4
    for (int r = 0; r < MT; ++r) m = fmaxf(m, (float)bufA[r * STRA + n]);
    atomicMax((int*)(gmax + (size_t)sample * 512 + n), __float_as_int(m));
  }
}

// ---------------------------------------------------------------------------
// Dense head: 512 ->256 ->128 -> outDim (LN+ReLU between), optional L2-norm.
// One block per sample row (tiny FLOPs; scalar is fine).
// ---------------------------------------------------------------------------
__global__ __launch_bounds__(256)
void k_head(const float* __restrict__ gin,
  const float* __restrict__ dw0, const float* __restrict__ db0,
  const float* __restrict__ dg0, const float* __restrict__ dbe0,
  const float* __restrict__ dw1, const float* __restrict__ db1,
  const float* __restrict__ dg1, const float* __restrict__ dbe1,
  const float* __restrict__ ow,  const float* __restrict__ ob,
  int outDim, int normalize, float* __restrict__ outp)
{
  __shared__ float xs[512];
  __shared__ float red[256];
  const int t = threadIdx.x;
  const int row = blockIdx.x;
  xs[t]       = gin[(size_t)row * 512 + t];
  xs[t + 256] = gin[(size_t)row * 512 + t + 256];
  __syncthreads();

  // 512 -> 256
  float y = db0[t];
  for (int k = 0; k < 512; ++k) y += xs[k] * dw0[k * 256 + t];
  float s  = block_sum(y, red);
  float ss = block_sum(y * y, red);
  float mean = s * (1.f / 256.f), var = ss * (1.f / 256.f) - mean * mean;
  float h = fmaxf((y - mean) * rsqrtf(var + LN_EPS) * dg0[t] + dbe0[t], 0.f);
  __syncthreads(); xs[t] = h; __syncthreads();

  // 256 -> 128
  float y1 = 0.f;
  if (t < 128) { y1 = db1[t]; for (int k = 0; k < 256; ++k) y1 += xs[k] * dw1[k * 128 + t]; }
  s  = block_sum(t < 128 ? y1 : 0.f, red);
  ss = block_sum(t < 128 ? y1 * y1 : 0.f, red);
  mean = s * (1.f / 128.f); var = ss * (1.f / 128.f) - mean * mean;
  const int tc = (t < 128) ? t : 0;
  float h1 = fmaxf((y1 - mean) * rsqrtf(var + LN_EPS) * dg1[tc] + dbe1[tc], 0.f);
  __syncthreads(); if (t < 128) xs[t] = h1; __syncthreads();

  // 128 -> outDim
  float o = 0.f;
  if (t < outDim) { o = ob[t]; for (int k = 0; k < 128; ++k) o += xs[k] * ow[k * outDim + t]; }
  if (normalize) {
    float nn = block_sum(t < outDim ? o * o : 0.f, red);
    o *= rsqrtf(nn);
  }
  if (t < outDim) outp[(size_t)row * outDim + t] = o;
}

// ---------------------------------------------------------------------------
// CRC loss: sum over (r1<r2, b) of ||enc[r1,b] - enc[r2,b]||^2 / (6*32)
// ---------------------------------------------------------------------------
__global__ __launch_bounds__(256)
void k_crc(const float* __restrict__ enc, float* __restrict__ lossp)
{
  __shared__ float red[256];
  const int t = threadIdx.x;
  const int p0[6] = {0, 0, 0, 1, 1, 2};
  const int p1[6] = {1, 2, 3, 2, 3, 3};
  float s = 0.f;
  if (t < 192) {
    const int pr = t / 32, b = t % 32;
    const float* e1 = enc + (size_t)(p0[pr] * 32 + b) * 64;
    const float* e2 = enc + (size_t)(p1[pr] * 32 + b) * 64;
    for (int k = 0; k < 64; ++k) { float d = e1[k] - e2[k]; s += d * d; }
  }
  float tot = block_sum(s, red);
  if (t == 0) lossp[0] = tot * (1.f / 192.f);
}

// ---------------------------------------------------------------------------
// Decoder (64->16->32->36 LN/ReLU, 36->1536) + quaternion rotate -> d_out.
// One block per (r,b) row.
// ---------------------------------------------------------------------------
__device__ __forceinline__ void ln_relu_block(float y, int C,
    const float* __restrict__ gam, const float* __restrict__ bet,
    float* dst, float* red, int t)
{
  float s  = block_sum(t < C ? y : 0.f, red);
  float ss = block_sum(t < C ? y * y : 0.f, red);
  float mean = s / (float)C, var = ss / (float)C - mean * mean;
  if (t < C) dst[t] = fmaxf((y - mean) * rsqrtf(var + LN_EPS) * gam[t] + bet[t], 0.f);
  __syncthreads();
}

__global__ __launch_bounds__(256)
void k_decoder(const float* __restrict__ enc, const float* __restrict__ q,
  const float* __restrict__ w0, const float* __restrict__ b0,
  const float* __restrict__ g0, const float* __restrict__ be0,
  const float* __restrict__ w1, const float* __restrict__ b1,
  const float* __restrict__ g1, const float* __restrict__ be1,
  const float* __restrict__ w2, const float* __restrict__ b2,
  const float* __restrict__ g2, const float* __restrict__ be2,
  const float* __restrict__ w3, const float* __restrict__ b3,
  float* __restrict__ outp)
{
  __shared__ float zin[64];
  __shared__ float tmp[64];
  __shared__ float red[256];
  __shared__ float dec[1536];
  const int t = threadIdx.x;
  const int row = blockIdx.x;               // r*32 + b
  if (t < 64) zin[t] = enc[(size_t)row * 64 + t];
  __syncthreads();

  float y0 = 0.f;
  if (t < 16) { y0 = b0[t]; for (int k = 0; k < 64; ++k) y0 += zin[k] * w0[k * 16 + t]; }
  ln_relu_block(y0, 16, g0, be0, tmp, red, t);
  float y1 = 0.f;
  if (t < 32) { y1 = b1[t]; for (int k = 0; k < 16; ++k) y1 += tmp[k] * w1[k * 32 + t]; }
  ln_relu_block(y1, 32, g1, be1, tmp, red, t);
  float y2 = 0.f;
  if (t < 36) { y2 = b2[t]; for (int k = 0; k < 32; ++k) y2 += tmp[k] * w2[k * 36 + t]; }
  ln_relu_block(y2, 36, g2, be2, tmp, red, t);

  for (int j = t; j < 1536; j += 256) {
    float o = b3[j];
    for (int k = 0; k < 36; ++k) o += tmp[k] * w3[k * 1536 + j];
    dec[j] = o;
  }
  __syncthreads();

  const float* qb = q + (size_t)(row & 31) * 4;
  const float qx = qb[0], qy = qb[1], qz = qb[2], qw = qb[3];
  for (int n = t; n < 512; n += 256) {
    const float px = dec[n * 3], py = dec[n * 3 + 1], pz = dec[n * 3 + 2];
    const float tx = 2.f * (qy * pz - qz * py);
    const float ty = 2.f * (qz * px - qx * pz);
    const float tz = 2.f * (qx * py - qy * px);
    const size_t o = (size_t)row * 1536 + (size_t)n * 3;
    outp[o]     = px + qw * tx + (qy * tz - qz * ty);
    outp[o + 1] = py + qw * ty + (qz * tx - qx * tz);
    outp[o + 2] = pz + qw * tz + (qx * ty - qy * tx);
  }
}

// ---------------------------------------------------------------------------
// Overlap loss: per (r,b), min pairwise distance per point (diag := 2R),
// mean of (2R - min(mindis, 2R)) over (R,B,N). 2R = 1.0.
// ---------------------------------------------------------------------------
__global__ __launch_bounds__(256)
void k_overlap(const float* __restrict__ rot, float* __restrict__ lossp)
{
  __shared__ float px[512], py[512], pz[512], sq[512];
  __shared__ float red[256];
  const int t = threadIdx.x;
  const float* base = rot + (size_t)blockIdx.x * 1536;
  for (int n = t; n < 512; n += 256) {
    const float x = base[n * 3], y = base[n * 3 + 1], z = base[n * 3 + 2];
    px[n] = x; py[n] = y; pz[n] = z; sq[n] = x * x + y * y + z * z;
  }
  __syncthreads();
  float local = 0.f;
  for (int i = t; i < 512; i += 256) {
    const float xi = px[i], yi = py[i], zi = pz[i], si = sq[i];
    float minv = 1.0f;                                  // two_r
    for (int j = 0; j < 512; ++j) {
      const float d = si + sq[j] - 2.f * (xi * px[j] + yi * py[j] + zi * pz[j]);
      minv = (j == i) ? minv : fminf(minv, d);
    }
    local += 1.0f - minv;                               // minv <= two_r
  }
  const float s = block_sum(local, red);
  if (t == 0) atomicAdd(lossp, s * (1.f / 65536.f));    // /(R*B*N)
}

// ---------------------------------------------------------------------------
// Utility kernels
// ---------------------------------------------------------------------------
__global__ void k_init(float* __restrict__ gt, float* __restrict__ ge,
                       float* __restrict__ loss2)
{
  const int i = blockIdx.x * 256 + threadIdx.x;
  if (i < 16384) gt[i] = 0.f;
  if (i < 65536) ge[i] = 0.f;
  if (i < 2)     loss2[i] = 0.f;
}

// W (K x N, f32, row-major) -> Wt (N x K, f16) : K-contiguous B layout.
__global__ void k_w2h(const float* __restrict__ W, _Float16* __restrict__ Wt,
                      int K, int N)
{
  const int i = blockIdx.x * 256 + threadIdx.x;
  if (i < K * N) {
    const int k = i / N, n = i - k * N;
    Wt[(size_t)n * K + k] = (_Float16)W[i];
  }
}

// pc_rot[r][b][n][i] = sum_j RM[r][i][j] * pc[b][n][j]
__global__ void k_rotate(const float* __restrict__ pc,
                         const float* __restrict__ rm,
                         float* __restrict__ outp)
{
  const int i = blockIdx.x * 256 + threadIdx.x;   // over R*B*N = 65536
  if (i >= 65536) return;
  const int r  = i >> 14;
  const int bn = i & 16383;
  const float* p = pc + (size_t)bn * 3;
  const float* m = rm + r * 9;
  const float x = p[0], y = p[1], z = p[2];
  float* o = outp + (size_t)i * 3;
  o[0] = m[0] * x + m[1] * y + m[2] * z;
  o[1] = m[3] * x + m[4] * y + m[5] * z;
  o[2] = m[6] * x + m[7] * y + m[8] * z;
}

// ---------------------------------------------------------------------------
// Host-side launch. Input order = setup_inputs() insertion order, nested
// dicts flattened in insertion order:
//   0 point_cloud, 1 rotate_matrix,
//   2..27  t_params {cw0,cb0,cg0,cbe0, cw1.., cw2.., cw3.., dw0.., dw1.., ow, ob}
//   28..53 e_params (same layout),
//   54..67 d_params {w0,b0,g0,be0, w1.., w2.., w3, b3}
// d_out: rotated (196608 f32) + crc_loss + overlap_loss.
// ---------------------------------------------------------------------------
#define IN(i) ((const float*)d_in[(i)])

extern "C" void kernel_launch(void* const* d_in, const int* in_sizes, int n_in,
                              void* d_out, int out_size, void* d_ws, size_t ws_size,
                              hipStream_t stream)
{
  (void)in_sizes; (void)n_in; (void)out_size; (void)ws_size;
  float* out = (float*)d_out;
  char*  ws  = (char*)d_ws;

  size_t off = 0;
  auto alloc = [&](size_t bytes) -> char* {
    off = (off + 255) & ~(size_t)255;
    char* p = ws + off;
    off += bytes;
    return p;
  };

  _Float16* wtT1 = (_Float16*)alloc((size_t)128 * 64 * 2);
  _Float16* wtT2 = (_Float16*)alloc((size_t)256 * 128 * 2);
  _Float16* wtT3 = (_Float16*)alloc((size_t)512 * 256 * 2);
  _Float16* wtE1 = (_Float16*)alloc((size_t)128 * 64 * 2);
  _Float16* wtE2 = (_Float16*)alloc((size_t)256 * 128 * 2);
  _Float16* wtE3 = (_Float16*)alloc((size_t)512 * 256 * 2);
  float* pcRot = (float*)alloc((size_t)65536 * 3 * 4);
  float* gT    = (float*)alloc((size_t)16384 * 4);
  float* gE    = (float*)alloc((size_t)65536 * 4);
  float* qbuf  = (float*)alloc((size_t)32 * 4 * 4);
  float* encb  = (float*)alloc((size_t)128 * 64 * 4);

  const float* pc = IN(0);
  const float* rm = IN(1);

  k_init<<<256, 256, 0, stream>>>(gT, gE, out + 196608);

  k_w2h<<<(64 * 128 + 255) / 256, 256, 0, stream>>>(IN(6),  wtT1, 64, 128);
  k_w2h<<<(128 * 256 + 255) / 256, 256, 0, stream>>>(IN(10), wtT2, 128, 256);
  k_w2h<<<(256 * 512 + 255) / 256, 256, 0, stream>>>(IN(14), wtT3, 256, 512);
  k_w2h<<<(64 * 128 + 255) / 256, 256, 0, stream>>>(IN(32), wtE1, 64, 128);
  k_w2h<<<(128 * 256 + 255) / 256, 256, 0, stream>>>(IN(36), wtE2, 128, 256);
  k_w2h<<<(256 * 512 + 255) / 256, 256, 0, stream>>>(IN(40), wtE3, 256, 512);

  k_rotate<<<65536 / 256, 256, 0, stream>>>(pc, rm, pcRot);

  // t-backbone over 16384 points
  k_backbone<<<16384 / MT, 256, LDS_BYTES, stream>>>(pc,
      IN(2), IN(3), IN(4), IN(5),
      wtT1, IN(7),  IN(8),  IN(9),
      wtT2, IN(11), IN(12), IN(13),
      wtT3, IN(15), IN(16), IN(17), gT);

  // e-backbone over 65536 rotated points
  k_backbone<<<65536 / MT, 256, LDS_BYTES, stream>>>(pcRot,
      IN(28), IN(29), IN(30), IN(31),
      wtE1, IN(33), IN(34), IN(35),
      wtE2, IN(37), IN(38), IN(39),
      wtE3, IN(41), IN(42), IN(43), gE);

  // heads: t -> q (normalized, 32x4); e -> enc (128x64)
  k_head<<<32, 256, 0, stream>>>(gT,
      IN(18), IN(19), IN(20), IN(21),
      IN(22), IN(23), IN(24), IN(25),
      IN(26), IN(27), 4, 1, qbuf);
  k_head<<<128, 256, 0, stream>>>(gE,
      IN(44), IN(45), IN(46), IN(47),
      IN(48), IN(49), IN(50), IN(51),
      IN(52), IN(53), 64, 0, encb);

  k_crc<<<1, 256, 0, stream>>>(encb, out + 196608);

  k_decoder<<<128, 256, 0, stream>>>(encb, qbuf,
      IN(54), IN(55), IN(56), IN(57),
      IN(58), IN(59), IN(60), IN(61),
      IN(62), IN(63), IN(64), IN(65),
      IN(66), IN(67), out);

  k_overlap<<<128, 256, 0, stream>>>(out, out + 196609);
}
